// GATv2_11493332484783
// MI455X (gfx1250) — matrix operand order repaired and checked by reading, not compiled, not run on hardware
//
#include <hip/hip_runtime.h>
#include <hip/hip_bf16.h>

// ---------------- static problem config (matches reference) ----------------
constexpr int B_   = 192;
constexpr int NPG_ = 48;
constexpr int K_   = 24;           // top-k keep
constexpr int N_   = B_ * NPG_;    // 9216 nodes
constexpr int DEG_ = 8;
constexpr int E0_  = N_ * DEG_;    // 73728 input edges
constexpr int E_   = E0_ + N_;     // + self loops = 82944
constexpr int FIN_ = 64;
constexpr int H_   = 8;
constexpr int C_   = 128;
constexpr int HC_  = H_ * C_;      // 1024
constexpr float NEG_ = 0.2f;

typedef __attribute__((ext_vector_type(16))) _Float16 v16h;
typedef __attribute__((ext_vector_type(8)))  _Float16 v8h;
typedef __attribute__((ext_vector_type(8)))  float    v8f;

// ------------------------- small device helpers ---------------------------
__device__ __forceinline__ unsigned encf(float f) {
    unsigned u = __float_as_uint(f);
    return (u & 0x80000000u) ? ~u : (u | 0x80000000u);
}
__device__ __forceinline__ float decf(unsigned e) {
    unsigned u = (e & 0x80000000u) ? (e & 0x7fffffffu) : ~e;
    return __uint_as_float(u);
}
__device__ __forceinline__ void edge_sd(const int* ei, int e, int& s, int& d) {
    if (e < E0_) { s = ei[e]; d = ei[E0_ + e]; }
    else         { s = d = e - E0_; }
}
__device__ __forceinline__ float leaky(float v) { return v >= 0.f ? v : NEG_ * v; }

// ------------------------------ utilities ---------------------------------
__global__ void k_cast_f2h(const float* __restrict__ in, _Float16* __restrict__ out, int n) {
    int i = blockIdx.x * blockDim.x + threadIdx.x;
    if (i < n) out[i] = (_Float16)in[i];
}
__global__ void k_fill_u32(unsigned* p, unsigned v, int n) {
    int i = blockIdx.x * blockDim.x + threadIdx.x;
    if (i < n) p[i] = v;
}
__global__ void k_fill_f32(float* p, float v, int n) {
    int i = blockIdx.x * blockDim.x + threadIdx.x;
    if (i < n) p[i] = v;
}

// -------- fuse fp32->fp16 cast with repack of B into WMMA fragment order ---
// Packed element i = (((kt*tilesN + nt)*32 + lane)*16 + j) holds
// B[kt*32 + (lane>>4)*16 + j][nt*16 + (lane&15)]  (B row-major [K,Nc]).
__global__ void k_pack_b(const float* __restrict__ B, _Float16* __restrict__ Bp,
                         int Kd, int Nc)
{
    const int i = blockIdx.x * blockDim.x + threadIdx.x;
    if (i >= Kd * Nc) return;
    const int j    = i & 15;
    const int lane = (i >> 4) & 31;
    const int tile = i >> 9;                 // / (32*16)
    const int tilesN = Nc >> 4;
    const int kt = tile / tilesN, nt = tile % tilesN;
    const int k = kt * 32 + (lane >> 4) * 16 + j;
    const int n = nt * 16 + (lane & 15);
    Bp[i] = (_Float16)B[(size_t)k * Nc + n];
}

// ------------------------------ WMMA GEMM ---------------------------------
// C[M,Nc] = A[M,K](f16 row-major) * Bp(pre-packed fragments) + bias
// act: 0 = identity, 1 = ELU.  One wave per 16x16 tile.
// Inner loop: 4x global_load_b128 + prefetch + one v_wmma.
__global__ void k_wmma_gemm(const _Float16* __restrict__ A,
                            const _Float16* __restrict__ Bp,
                            const float*    __restrict__ bias,
                            float*          __restrict__ outF,
                            _Float16*       __restrict__ outH,
                            int M, int Nc, int Kd, int act)
{
    const int wave = threadIdx.x >> 5;
    const int lane = threadIdx.x & 31;
    const int tilesN = Nc >> 4;
    const int totalTiles = (M >> 4) * tilesN;
    const int tile = blockIdx.x * (blockDim.x >> 5) + wave;
    if (tile >= totalTiles) return;                 // wave-uniform: EXEC stays full
    const int tm = tile / tilesN;
    const int tn = tile % tilesN;
    const int l  = lane & 15;
    const int g  = lane >> 4;

    // A fragment: lane owns row tm*16+l; elements [0..15]=K g*16..g*16+15
    const _Float16* ap = A + (size_t)(tm * 16 + l) * Kd + g * 16;
    // Packed B fragment for this lane, kt = 0
    const _Float16* bp = Bp + ((size_t)tn * 32 + lane) * 16;
    const size_t bstep = (size_t)tilesN * 512;      // halves per kt step

    v8f c = {};
    for (int k0 = 0; k0 < Kd; k0 += 32) {
        __builtin_prefetch(ap + 32, 0, 0);          // -> global_prefetch_b8
        __builtin_prefetch(bp + bstep, 0, 0);
        v8h alo = *(const v8h*)(ap);
        v8h ahi = *(const v8h*)(ap + 8);
        v8h blo = *(const v8h*)(bp);
        v8h bhi = *(const v8h*)(bp + 8);
        v16h a, b;
#pragma unroll
        for (int i = 0; i < 8; ++i) {
            a[i] = alo[i]; a[i + 8] = ahi[i];
            b[i] = blo[i]; b[i + 8] = bhi[i];
        }
        c = __builtin_amdgcn_wmma_f32_16x16x32_f16(
                false, a, false, b, (short)0, c, false, false);
        ap += 32;
        bp += bstep;
    }

    const float bv = bias ? bias[tn * 16 + l] : 0.0f;
#pragma unroll
    for (int i = 0; i < 8; ++i) {
        float v = c[i] + bv;
        if (act == 1) v = v > 0.f ? v : (__expf(v) - 1.f);   // ELU
        const int r = tm * 16 + g * 8 + i;                   // C/D layout
        const size_t o = (size_t)r * Nc + tn * 16 + l;
        if (outF) outF[o] = v;
        if (outH) outH[o] = (_Float16)v;
    }
}

// -------------------------- GATv2 edge stages -----------------------------
__global__ void k_edge_logits(const _Float16* __restrict__ xl,
                              const _Float16* __restrict__ xr,
                              const float* __restrict__ att,
                              const int* __restrict__ ei,
                              float* __restrict__ logit,
                              unsigned* __restrict__ maxenc)
{
    const int e = blockIdx.x;
    int s, d; edge_sd(ei, e, s, d);
    __shared__ float hsum[H_];
    if (threadIdx.x < H_) hsum[threadIdx.x] = 0.f;
    __syncthreads();
    const int t = threadIdx.x;                   // 128 threads
#pragma unroll
    for (int h = 0; h < H_; ++h) {
        const int idx = t + (h << 7);            // idx>>7 == h for t<128
        float v = (float)xl[(size_t)s * HC_ + idx] + (float)xr[(size_t)d * HC_ + idx];
        atomicAdd(&hsum[h], leaky(v) * att[idx]);
    }
    __syncthreads();
    if (t < H_) {
        const float v = hsum[t];
        logit[(size_t)e * H_ + t] = v;
        atomicMax(&maxenc[(size_t)d * H_ + t], encf(v));
    }
}

__global__ void k_edge_expsum(const float* __restrict__ logit,
                              const unsigned* __restrict__ maxenc,
                              const int* __restrict__ ei,
                              float* __restrict__ wbuf,
                              float* __restrict__ sumbuf)
{
    const int gi = blockIdx.x * blockDim.x + threadIdx.x;
    if (gi >= E_ * H_) return;
    const int e = gi >> 3, h = gi & 7;
    int s, d; edge_sd(ei, e, s, d); (void)s;
    const float w = __expf(logit[gi] - decf(maxenc[(size_t)d * H_ + h]));
    wbuf[gi] = w;
    atomicAdd(&sumbuf[(size_t)d * H_ + h], w);
}

__global__ void k_edge_aggregate(const _Float16* __restrict__ xl,
                                 const float* __restrict__ wbuf,
                                 const float* __restrict__ sumbuf,
                                 const int* __restrict__ ei,
                                 float* __restrict__ outF)
{
    const int e = blockIdx.x;
    int s, d; edge_sd(ei, e, s, d);
    for (int idx = threadIdx.x; idx < HC_; idx += blockDim.x) {
        const int h = idx >> 7;
        const float alpha = wbuf[(size_t)e * H_ + h] /
                            (sumbuf[(size_t)d * H_ + h] + 1e-16f);
        atomicAdd(&outF[(size_t)d * HC_ + idx],
                  alpha * (float)xl[(size_t)s * HC_ + idx]);
    }
}

// --------------------------- ASAP pooling stages ---------------------------
__global__ void k_segmax_x1(const float* __restrict__ x1,
                            const int* __restrict__ ei,
                            unsigned* __restrict__ menc)
{
    const int e = blockIdx.x;
    int s, d; edge_sd(ei, e, s, d);
    const int f = threadIdx.x;                   // 128
    atomicMax(&menc[(size_t)d * 128 + f], encf(x1[(size_t)s * 128 + f]));
}

__global__ void k_decode_max(const unsigned* __restrict__ menc,
                             _Float16* __restrict__ outH, int n)
{
    const int i = blockIdx.x * blockDim.x + threadIdx.x;
    if (i < n) outH[i] = (_Float16)decf(menc[i]);
}

__global__ void k_node_q(const float* __restrict__ xq,
                         const float* __restrict__ x1,
                         const float* __restrict__ attW,
                         float* __restrict__ q1, float* __restrict__ q2)
{
    const int i = blockIdx.x * blockDim.x + threadIdx.x;
    if (i >= N_) return;
    float s1 = 0.f, s2 = 0.f;
    for (int f = 0; f < 128; ++f) {
        s1 += xq[(size_t)i * 128 + f] * attW[f];
        s2 += x1[(size_t)i * 128 + f] * attW[128 + f];
    }
    q1[i] = s1; q2[i] = s2;
}

__global__ void k_edge_sc(const float* __restrict__ q1,
                          const float* __restrict__ q2,
                          const float* __restrict__ attb,
                          const int* __restrict__ ei,
                          float* __restrict__ scl,
                          unsigned* __restrict__ scmax)
{
    const int e = blockIdx.x * blockDim.x + threadIdx.x;
    if (e >= E_) return;
    int s, d; edge_sd(ei, e, s, d);
    const float v = leaky(q1[d] + q2[s] + attb[0]);
    scl[e] = v;
    atomicMax(&scmax[d], encf(v));
}

__global__ void k_edge_scexp(const float* __restrict__ scl,
                             const unsigned* __restrict__ scmax,
                             const int* __restrict__ ei,
                             float* __restrict__ scw,
                             float* __restrict__ scsum)
{
    const int e = blockIdx.x * blockDim.x + threadIdx.x;
    if (e >= E_) return;
    int s, d; edge_sd(ei, e, s, d); (void)s;
    const float w = __expf(scl[e] - decf(scmax[d]));
    scw[e] = w;
    atomicAdd(&scsum[d], w);
}

__global__ void k_edge_xc(const float* __restrict__ x1,
                          const float* __restrict__ scw,
                          const float* __restrict__ scsum,
                          const int* __restrict__ ei,
                          float* __restrict__ xc)
{
    const int e = blockIdx.x;
    int s, d; edge_sd(ei, e, s, d);
    const int f = threadIdx.x;                   // 128
    const float alpha = scw[e] / (scsum[d] + 1e-16f);
    atomicAdd(&xc[(size_t)d * 128 + f], alpha * x1[(size_t)s * 128 + f]);
}

// ------------------------------ LEConv / fit -------------------------------
__global__ void k_node_le(const float* __restrict__ xc,
                          const float* __restrict__ le1W, const float* __restrict__ le1b,
                          const float* __restrict__ le2W,
                          const float* __restrict__ le3W, const float* __restrict__ le3b,
                          float* __restrict__ av, float* __restrict__ bv,
                          float* __restrict__ cv)
{
    const int i = blockIdx.x * blockDim.x + threadIdx.x;
    if (i >= N_) return;
    float a = 0.f, b = 0.f, c = 0.f;
    for (int f = 0; f < 128; ++f) {
        const float x = xc[(size_t)i * 128 + f];
        a += x * le1W[f]; b += x * le2W[f]; c += x * le3W[f];
    }
    av[i] = a + le1b[0];
    bv[i] = b;
    cv[i] = c + le3b[0];
}

__global__ void k_edge_fit(const float* __restrict__ av,
                           const float* __restrict__ bv,
                           const int* __restrict__ ei,
                           float* __restrict__ fitacc)
{
    const int e = blockIdx.x * blockDim.x + threadIdx.x;
    if (e >= E_) return;
    int s, d; edge_sd(ei, e, s, d);
    atomicAdd(&fitacc[d], av[s] - bv[d]);
}

__global__ void k_node_fit(const float* __restrict__ fitacc,
                           const float* __restrict__ cv,
                           float* __restrict__ fit)
{
    const int i = blockIdx.x * blockDim.x + threadIdx.x;
    if (i >= N_) return;
    const float z = fitacc[i] + cv[i];
    fit[i] = 1.f / (1.f + __expf(-z));
}

// ------------------------- top-k + mean pool (per graph) -------------------
__global__ void k_topk_pool(const float* __restrict__ fit,
                            const float* __restrict__ xc,
                            float* __restrict__ gbuf)
{
    const int gi = blockIdx.x;
    __shared__ float fs[NPG_];
    __shared__ int   sel[NPG_];
    const int t = threadIdx.x;                    // 128
    if (t < NPG_) fs[t] = fit[gi * NPG_ + t];
    __syncthreads();
    if (t < NPG_) {
        int rank = 0;
        for (int j = 0; j < NPG_; ++j)
            rank += (fs[j] > fs[t]) || (fs[j] == fs[t] && j < t);
        sel[t] = (rank < K_);
    }
    __syncthreads();
    float acc = 0.f;
    for (int j = 0; j < NPG_; ++j)
        if (sel[j]) acc += xc[(size_t)(gi * NPG_ + j) * 128 + t] * fs[j];
    gbuf[(size_t)gi * 128 + t] = acc * (1.0f / K_);
}

// ------------------------------- MLP head ----------------------------------
__global__ void k_mlp(const float* __restrict__ gbuf,
                      const float* __restrict__ f1W, const float* __restrict__ f1b,
                      const float* __restrict__ f2W, const float* __restrict__ f2b,
                      const float* __restrict__ f3W, const float* __restrict__ f3b,
                      float* __restrict__ out)
{
    const int gi = blockIdx.x;
    const int t = threadIdx.x;                    // 128
    __shared__ float gs[128], h1[16], h2[32];
    gs[t] = gbuf[(size_t)gi * 128 + t];
    __syncthreads();
    if (t < 16) {
        float a = f1b[t];
        for (int i = 0; i < 128; ++i) a += gs[i] * f1W[i * 16 + t];
        h1[t] = fmaxf(a, 0.f);
    }
    __syncthreads();
    if (t < 32) {
        float a = f2b[t];
        for (int i = 0; i < 16; ++i) a += h1[i] * f2W[i * 32 + t];
        h2[t] = fmaxf(a, 0.f);
    }
    __syncthreads();
    if (t < 5) {
        float a = f3b[t];
        for (int i = 0; i < 32; ++i) a += h2[i] * f3W[i * 5 + t];
        out[gi * 5 + t] = a;
    }
}

// ------------------------------- launcher ----------------------------------
static inline int ceil_div(int a, int b) { return (a + b - 1) / b; }

extern "C" void kernel_launch(void* const* d_in, const int* in_sizes, int n_in,
                              void* d_out, int out_size, void* d_ws, size_t ws_size,
                              hipStream_t stream)
{
    (void)in_sizes; (void)n_in; (void)out_size; (void)ws_size;

    // ---- inputs (setup_inputs order) ----
    const float* x      = (const float*)d_in[0];
    const int*   ei     = (const int*)  d_in[1];
    const float* g_Wl   = (const float*)d_in[2];
    const float* g_bl   = (const float*)d_in[3];
    const float* g_Wr   = (const float*)d_in[4];
    const float* g_br   = (const float*)d_in[5];
    const float* g_att  = (const float*)d_in[6];
    const float* g_Wres = (const float*)d_in[7];
    const float* g_bias = (const float*)d_in[8];
    const float* l_W    = (const float*)d_in[9];
    const float* l_b    = (const float*)d_in[10];
    const float* plinW  = (const float*)d_in[11];
    const float* plinb  = (const float*)d_in[12];
    const float* pattW  = (const float*)d_in[13];
    const float* pattb  = (const float*)d_in[14];
    const float* le1W   = (const float*)d_in[15];
    const float* le1b   = (const float*)d_in[16];
    const float* le2W   = (const float*)d_in[17];
    const float* le3W   = (const float*)d_in[18];
    const float* le3b   = (const float*)d_in[19];
    const float* f1W    = (const float*)d_in[20];
    const float* f1b    = (const float*)d_in[21];
    const float* f2W    = (const float*)d_in[22];
    const float* f2b    = (const float*)d_in[23];
    const float* f3W    = (const float*)d_in[24];
    const float* f3b    = (const float*)d_in[25];
    float* out = (float*)d_out;

    // ---- workspace bump allocator (256B aligned) ----
    char* base = (char*)d_ws;
    size_t off = 0;
    auto alloc = [&](size_t bytes) -> void* {
        void* p = base + off;
        off = (off + bytes + 255) & ~(size_t)255;
        return p;
    };
    _Float16* xh       = (_Float16*)alloc((size_t)N_ * FIN_ * 2);
    _Float16* WlP      = (_Float16*)alloc((size_t)FIN_ * HC_ * 2);   // packed fragments
    _Float16* WrP      = (_Float16*)alloc((size_t)FIN_ * HC_ * 2);
    _Float16* WresP    = (_Float16*)alloc((size_t)FIN_ * HC_ * 2);
    _Float16* lWP      = (_Float16*)alloc((size_t)HC_ * 128 * 2);
    _Float16* plinWP   = (_Float16*)alloc((size_t)128 * 128 * 2);
    _Float16* xlH      = (_Float16*)alloc((size_t)N_ * HC_ * 2);
    _Float16* xrH      = (_Float16*)alloc((size_t)N_ * HC_ * 2);
    float*    outF     = (float*)   alloc((size_t)N_ * HC_ * 4);
    _Float16* outH     = (_Float16*)alloc((size_t)N_ * HC_ * 2);
    float*    logit    = (float*)   alloc((size_t)E_ * H_ * 4);
    float*    wbuf     = (float*)   alloc((size_t)E_ * H_ * 4);
    unsigned* maxenc   = (unsigned*)alloc((size_t)N_ * H_ * 4);
    float*    sumbuf   = (float*)   alloc((size_t)N_ * H_ * 4);
    float*    x1F      = (float*)   alloc((size_t)N_ * 128 * 4);
    unsigned* x1maxenc = (unsigned*)alloc((size_t)N_ * 128 * 4);
    _Float16* x1maxH   = (_Float16*)alloc((size_t)N_ * 128 * 2);
    float*    xqF      = (float*)   alloc((size_t)N_ * 128 * 4);
    float*    q1       = (float*)   alloc((size_t)N_ * 4);
    float*    q2       = (float*)   alloc((size_t)N_ * 4);
    float*    scl      = (float*)   alloc((size_t)E_ * 4);
    float*    scw      = (float*)   alloc((size_t)E_ * 4);
    unsigned* scmax    = (unsigned*)alloc((size_t)N_ * 4);
    float*    scsum    = (float*)   alloc((size_t)N_ * 4);
    float*    xcF      = (float*)   alloc((size_t)N_ * 128 * 4);
    float*    av       = (float*)   alloc((size_t)N_ * 4);
    float*    bv       = (float*)   alloc((size_t)N_ * 4);
    float*    cv       = (float*)   alloc((size_t)N_ * 4);
    float*    fitacc   = (float*)   alloc((size_t)N_ * 4);
    float*    fitv     = (float*)   alloc((size_t)N_ * 4);
    float*    gbuf     = (float*)   alloc((size_t)B_ * 128 * 4);

    const int T = 256;
    auto pack = [&](const float* src, _Float16* dst, int Kd, int Nc) {
        k_pack_b<<<ceil_div(Kd * Nc, T), T, 0, stream>>>(src, dst, Kd, Nc);
    };
    auto gemm = [&](const _Float16* A, const _Float16* Bp, const float* bias,
                    float* oF, _Float16* oH, int M, int Nc, int Kd, int act) {
        const int tiles = (M / 16) * (Nc / 16);     // 8 waves per 256-thread block
        k_wmma_gemm<<<ceil_div(tiles, 8), 256, 0, stream>>>(A, Bp, bias, oF, oH, M, Nc, Kd, act);
    };

    // ---- fp16 staging: activations cast, weights cast+packed to fragments ----
    k_cast_f2h<<<ceil_div(N_ * FIN_, T), T, 0, stream>>>(x, xh, N_ * FIN_);
    pack(g_Wl,   WlP,    FIN_, HC_);
    pack(g_Wr,   WrP,    FIN_, HC_);
    pack(g_Wres, WresP,  FIN_, HC_);
    pack(l_W,    lWP,    HC_,  128);
    pack(plinW,  plinWP, 128,  128);

    // ---- GATv2 linear projections (WMMA) ----
    gemm(xh, WlP,   g_bl,   nullptr, xlH, N_, HC_, FIN_, 0);
    gemm(xh, WrP,   g_br,   nullptr, xrH, N_, HC_, FIN_, 0);
    gemm(xh, WresP, g_bias, outF,    nullptr, N_, HC_, FIN_, 0);   // residual + bias seeds accumulator

    // ---- edge attention + softmax + aggregation ----
    k_fill_u32<<<ceil_div(N_ * H_, T), T, 0, stream>>>(maxenc, 0u, N_ * H_);
    k_fill_f32<<<ceil_div(N_ * H_, T), T, 0, stream>>>(sumbuf, 0.f, N_ * H_);
    k_edge_logits<<<E_, 128, 0, stream>>>(xlH, xrH, g_att, ei, logit, maxenc);
    k_edge_expsum<<<ceil_div(E_ * H_, T), T, 0, stream>>>(logit, maxenc, ei, wbuf, sumbuf);
    k_edge_aggregate<<<E_, 256, 0, stream>>>(xlH, wbuf, sumbuf, ei, outF);

    // ---- x1 = elu(out @ l_W + l_b)  (WMMA with fused ELU) ----
    k_cast_f2h<<<ceil_div(N_ * HC_, T), T, 0, stream>>>(outF, outH, N_ * HC_);
    gemm(outH, lWP, l_b, x1F, nullptr, N_, 128, HC_, 1);

    // ---- ASAP: segment-max -> xq (WMMA) ----
    k_fill_u32<<<ceil_div(N_ * 128, T), T, 0, stream>>>(x1maxenc, 0u, N_ * 128);
    k_segmax_x1<<<E_, 128, 0, stream>>>(x1F, ei, x1maxenc);
    k_decode_max<<<ceil_div(N_ * 128, T), T, 0, stream>>>(x1maxenc, x1maxH, N_ * 128);
    gemm(x1maxH, plinWP, plinb, xqF, nullptr, N_, 128, 128, 0);

    // ---- ASAP attention scores + weighted cluster features ----
    k_node_q<<<ceil_div(N_, T), T, 0, stream>>>(xqF, x1F, pattW, q1, q2);
    k_fill_u32<<<ceil_div(N_, T), T, 0, stream>>>(scmax, 0u, N_);
    k_fill_f32<<<ceil_div(N_, T), T, 0, stream>>>(scsum, 0.f, N_);
    k_edge_sc<<<ceil_div(E_, T), T, 0, stream>>>(q1, q2, pattb, ei, scl, scmax);
    k_edge_scexp<<<ceil_div(E_, T), T, 0, stream>>>(scl, scmax, ei, scw, scsum);
    k_fill_f32<<<ceil_div(N_ * 128, T), T, 0, stream>>>(xcF, 0.f, N_ * 128);
    k_edge_xc<<<E_, 128, 0, stream>>>(x1F, scw, scsum, ei, xcF);

    // ---- LEConv fitness ----
    k_node_le<<<ceil_div(N_, T), T, 0, stream>>>(xcF, le1W, le1b, le2W, le3W, le3b, av, bv, cv);
    k_fill_f32<<<ceil_div(N_, T), T, 0, stream>>>(fitacc, 0.f, N_);
    k_edge_fit<<<ceil_div(E_, T), T, 0, stream>>>(av, bv, ei, fitacc);
    k_node_fit<<<ceil_div(N_, T), T, 0, stream>>>(fitacc, cv, fitv);

    // ---- per-graph top-k + mean pool + MLP head ----
    k_topk_pool<<<B_, 128, 0, stream>>>(fitv, xcF, gbuf);
    k_mlp<<<B_, 128, 0, stream>>>(gbuf, f1W, f1b, f2W, f2b, f3W, f3b, out);
}
